// PairStr2Pair_42356967473473
// MI455X (gfx1250) — matrix operand order repaired and checked by reading, not compiled, not run on hardware
//
#include <hip/hip_runtime.h>

// ---------------------------------------------------------------------------
// PairStr2Pair block for MI455X (gfx1250, wave32).
//  * Heavy einsums on v_wmma_f32_16x16x32_bf16 (f32 accumulate), operands
//    staged in bf16 in WMMA-native layouts (fragments = contiguous b128).
//  * 1x4 N-blocking: each wave owns 4 accumulator tiles, reusing A fragments
//    4x and keeping 4 independent WMMA chains in flight.
//  * k_tri_mm uses the Tensor Data Mover: double-buffered
//    tensor_load_to_lds of 16x64 bf16 panels + s_wait_tensorcnt, WMMA
//    fragments read from LDS via ds_load_b128.
//  * ws requirement: ~413 MB.
// ---------------------------------------------------------------------------

#define L   512
#define DP  128
#define DH  32
#define HN  4
#define HST 16
#define DRBF 64
#define DST 32
#define LL  (L*L)

typedef __attribute__((ext_vector_type(16))) __bf16        v16bf;
typedef __attribute__((ext_vector_type(8)))  float         v8f;
typedef __attribute__((ext_vector_type(4)))  unsigned int  v4u;
typedef __attribute__((ext_vector_type(8)))  int           i32x8;
typedef __attribute__((ext_vector_type(4)))  int           i32x4;

union Frag { v4u q[2]; v16bf v; };

typedef unsigned short us;

#if defined(__gfx1250__) && __has_builtin(__builtin_amdgcn_tensor_load_to_lds)
#define HAVE_TDM 1
#else
#define HAVE_TDM 0
#endif

__device__ __forceinline__ us f2bf(float f) {
  unsigned int u = __float_as_uint(f);
  u += 0x7FFFu + ((u >> 16) & 1u);           // round-to-nearest-even
  return (us)(u >> 16);
}
__device__ __forceinline__ float bf2f(us h) {
  return __uint_as_float(((unsigned int)h) << 16);
}
__device__ __forceinline__ float sigm(float x) { return 1.0f / (1.0f + __expf(-x)); }

#if HAVE_TDM
// Flat->LDS byte offset: LDS aperture keeps the offset in addr[31:0].
__device__ __forceinline__ unsigned lds_off(const void* p) {
  return (unsigned)(size_t)p;
}
// One 2D tile move: 16 rows x 64 bf16, row stride 512 elements, into LDS.
// D# packed per CDNA5 ISA 8.3/8.4 (count=1, type=2, data_size=1 -> 2 bytes).
__device__ __forceinline__ void tdm_load_16x64(unsigned ldsaddr, const us* gsrc) {
  unsigned long long ga = (unsigned long long)(size_t)gsrc;
  v4u g0;
  g0[0] = 1u;                                        // count=1
  g0[1] = ldsaddr;                                   // lds_addr
  g0[2] = (unsigned)ga;                              // global_addr[31:0]
  g0[3] = (unsigned)((ga >> 32) & 0x01FFFFFFu) | (2u << 30);  // addr[56:32] | type=2
  i32x8 g1;
  g1[0] = 0x00010000;          // workgroup_mask=0, data_size=1 (2B)
  g1[1] = (int)(512u << 16);   // tensor_dim0 = 512 (low 16 in bits 63:48)
  g1[2] = (int)(512u << 16);   // tensor_dim1 = 512 (low 16 in bits 95:80)
  g1[3] = (int)(64u << 16);    // tile_dim0 = 64
  g1[4] = 16;                  // tile_dim1 = 16, tile_dim2 = 0
  g1[5] = 512;                 // tensor_dim0_stride = 512 (elements)
  g1[6] = 0;
  g1[7] = 0;
  i32x4 z; z[0] = z[1] = z[2] = z[3] = 0;
#if __clang_major__ >= 23
  i32x8 z8; for (int i = 0; i < 8; ++i) z8[i] = 0;
  __builtin_amdgcn_tensor_load_to_lds(g0, g1, z, z, z8, 0);
#else
  __builtin_amdgcn_tensor_load_to_lds(g0, g1, z, z, 0);
#endif
}
#endif

// ---------------------------------------------------------------------------
// Stage 0: state LN + left/right projections  (tiny)
// ---------------------------------------------------------------------------
__global__ void k_state_proj(const float* __restrict__ state,
                             const float* __restrict__ nsg, const float* __restrict__ nsb,
                             const float* __restrict__ plw, const float* __restrict__ plb,
                             const float* __restrict__ prw, const float* __restrict__ prb,
                             float* __restrict__ LR) {
  int l = blockIdx.x * blockDim.x + threadIdx.x;
  if (l >= L) return;
  float s[DST]; float m = 0.f;
  for (int c = 0; c < DST; ++c) { s[c] = state[l*DST + c]; m += s[c]; }
  m *= (1.0f / DST);
  float v = 0.f;
  for (int c = 0; c < DST; ++c) { float d = s[c] - m; v += d * d; }
  float inv = rsqrtf(v * (1.0f / DST) + 1e-5f);
  for (int c = 0; c < DST; ++c) s[c] = (s[c] - m) * inv * nsg[c] + nsb[c];
  for (int a = 0; a < HST; ++a) {
    float al = plb[a], ar = prb[a];
    for (int c = 0; c < DST; ++c) { al += s[c] * plw[c*HST + a]; ar += s[c] * prw[c*HST + a]; }
    LR[l*32 + a] = al; LR[l*32 + 16 + a] = ar;
  }
}

// ---------------------------------------------------------------------------
// Stage 1: bias tensor  gate(outer(left,right) @ to_gate) * (rbf @ emb_rbf)
// ---------------------------------------------------------------------------
__global__ void k_bias(const float* __restrict__ rbf_feat,
                       const float* __restrict__ LR,
                       const float* __restrict__ tgw, const float* __restrict__ tgb,
                       const float* __restrict__ erw, const float* __restrict__ erb,
                       float* __restrict__ BIAS) {
  int t = threadIdx.x;
  long pi = (long)blockIdx.x * 2 + (t >> 7);
  int p = t & 127;
  int i = (int)(pi >> 9), j = (int)(pi & 511);
  float lf[HST], rt[HST];
  for (int a = 0; a < HST; ++a) { lf[a] = LR[i*32 + a]; rt[a] = LR[j*32 + 16 + a]; }
  float gl = tgb[p];
  for (int a = 0; a < HST; ++a) {
    float la = lf[a];
    const float* w = tgw + (long)(a*16) * DP + p;
    for (int b = 0; b < HST; ++b) gl += la * rt[b] * w[(long)b * DP];
  }
  float rv = erb[p];
  const float* rf = rbf_feat + pi * DRBF;
  for (int r = 0; r < DRBF; ++r) rv += rf[r] * erw[r*DP + p];
  BIAS[pi*DP + p] = sigm(gl) * rv;
}

// ---------------------------------------------------------------------------
// Triangle multiplication
// ---------------------------------------------------------------------------
struct TriP {
  const float *ng,*nb,*lw,*lb,*rw,*rb,*lgw,*lgb,*rgw,*rgb,*gw,*gb,*nog,*nob,*ow,*ob;
};

__global__ void k_tri_pre(const float* __restrict__ pairbuf, TriP P,
                          us* __restrict__ Abuf, us* __restrict__ Bbuf, int transposed) {
  __shared__ float xs[4][DP];
  __shared__ float red[4][64];
  int t = threadIdx.x, sub = t >> 6, u = t & 63;
  long pi = (long)blockIdx.x * 4 + sub;
  int r = (int)(pi >> 9), c = (int)(pi & 511);
  const float* x = pairbuf + pi * DP;
  float v0 = x[u], v1 = x[u + 64];
  red[sub][u] = v0 + v1; __syncthreads();
  for (int s = 32; s > 0; s >>= 1) { if (u < s) red[sub][u] += red[sub][u+s]; __syncthreads(); }
  float mean = red[sub][0] * (1.0f / DP); __syncthreads();
  red[sub][u] = (v0-mean)*(v0-mean) + (v1-mean)*(v1-mean); __syncthreads();
  for (int s = 32; s > 0; s >>= 1) { if (u < s) red[sub][u] += red[sub][u+s]; __syncthreads(); }
  float inv = rsqrtf(red[sub][0] * (1.0f / DP) + 1e-5f); __syncthreads();
  xs[sub][u]      = (v0-mean)*inv*P.ng[u]    + P.nb[u];
  xs[sub][u + 64] = (v1-mean)*inv*P.ng[u+64] + P.nb[u+64];
  __syncthreads();
  int d = u & 31, half = u >> 5;
  const float* W  = half ? P.rw  : P.lw;
  const float* Wg = half ? P.rgw : P.lgw;
  const float* Bv = half ? P.rb  : P.lb;
  const float* Bg = half ? P.rgb : P.lgb;
  float a = 0.f, ag = 0.f;
  for (int ch = 0; ch < DP; ++ch) { float xv = xs[sub][ch]; a += xv * W[ch*DH + d]; ag += xv * Wg[ch*DH + d]; }
  float val = sigm(ag + Bg[d]) * (a + Bv[d]);
  long idx = transposed ? ((long)c * L + r) : ((long)r * L + c);
  us* dst = half ? Bbuf : Abuf;
  dst[(long)d * LL + idx] = f2bf(val);
}

// per-d 512x512x512 GEMM, 1x4 N-blocked.
#if HAVE_TDM
// TDM double-buffered: 5 tensor_load_to_lds per 64-wide K slab, WMMA reads LDS.
__global__ __launch_bounds__(32) void k_tri_mm(const us* __restrict__ A,
                                               const us* __restrict__ Bm,
                                               float* __restrict__ O) {
  __shared__ __align__(16) us sA[2][16*64];
  __shared__ __align__(16) us sB[4][2][16*64];
  int it = blockIdx.x, jq = blockIdx.y, d = blockIdx.z;
  int lane = threadIdx.x, sel = lane >> 4, mn = lane & 15;
  const us* Abase = A  + (long)d * LL + (long)(it*16) * L;
  const us* Bbase = Bm + (long)d * LL + (long)(jq*64) * L;
  v8f acc[4];
  for (int p = 0; p < 4; ++p) for (int r = 0; r < 8; ++r) acc[p][r] = 0.f;

  tdm_load_16x64(lds_off(&sA[0][0]), Abase);
  for (int p = 0; p < 4; ++p)
    tdm_load_16x64(lds_off(&sB[p][0][0]), Bbase + (long)p*16*L);

  for (int s = 0; s < 8; ++s) {
    int buf = s & 1;
    if (s < 7) {
      int kc = (s + 1) * 64;
      tdm_load_16x64(lds_off(&sA[buf^1][0]), Abase + kc);
      for (int p = 0; p < 4; ++p)
        tdm_load_16x64(lds_off(&sB[p][buf^1][0]), Bbase + (long)p*16*L + kc);
      __builtin_amdgcn_s_wait_tensorcnt(5);   // slab s complete, s+1 in flight
    } else {
      __builtin_amdgcn_s_wait_tensorcnt(0);
    }
    for (int kk = 0; kk < 64; kk += 32) {
      Frag a;
      const us* ap = &sA[buf][mn*64 + kk + sel*8];
      a.q[0] = *(const v4u*)ap;
      a.q[1] = *(const v4u*)(ap + 16);
      for (int p = 0; p < 4; ++p) {
        Frag b;
        const us* bp = &sB[p][buf][mn*64 + kk + sel*8];
        b.q[0] = *(const v4u*)bp;
        b.q[1] = *(const v4u*)(bp + 16);
        acc[p] = __builtin_amdgcn_wmma_f32_16x16x32_bf16(false, a.v, false, b.v,
                                                         (short)0, acc[p], false, false);
      }
    }
  }
  for (int p = 0; p < 4; ++p) {
    int j = jq*64 + p*16 + mn;
    for (int r = 0; r < 8; ++r) {
      int m = it*16 + r + sel*8;
      O[((long)m * L + j) * DH + d] = acc[p][r];
    }
  }
}
#else
__global__ __launch_bounds__(32) void k_tri_mm(const us* __restrict__ A,
                                               const us* __restrict__ Bm,
                                               float* __restrict__ O) {
  int it = blockIdx.x, jq = blockIdx.y, d = blockIdx.z;
  int lane = threadIdx.x, sel = lane >> 4, mn = lane & 15;
  const us* Ab = A + (long)d * LL + (long)(it*16 + mn) * L;
  const us* Bb[4];
  for (int p = 0; p < 4; ++p)
    Bb[p] = Bm + (long)d * LL + (long)(jq*64 + p*16 + mn) * L;
  v8f acc[4];
  for (int p = 0; p < 4; ++p) for (int r = 0; r < 8; ++r) acc[p][r] = 0.f;
  for (int kc = 0; kc < L; kc += 32) {
    Frag a;
    a.q[0] = *(const v4u*)(Ab + kc + sel*8);
    a.q[1] = *(const v4u*)(Ab + kc + 16 + sel*8);
    for (int p = 0; p < 4; ++p) {
      Frag b;
      b.q[0] = *(const v4u*)(Bb[p] + kc + sel*8);
      b.q[1] = *(const v4u*)(Bb[p] + kc + 16 + sel*8);
      acc[p] = __builtin_amdgcn_wmma_f32_16x16x32_bf16(false, a.v, false, b.v,
                                                       (short)0, acc[p], false, false);
    }
  }
  for (int p = 0; p < 4; ++p) {
    int j = jq*64 + p*16 + mn;
    for (int r = 0; r < 8; ++r) {
      int m = it*16 + r + sel*8;
      O[((long)m * L + j) * DH + d] = acc[p][r];
    }
  }
}
#endif

__global__ void k_tri_post(const float* __restrict__ xsrc,
                           const float* __restrict__ resid,
                           float* __restrict__ dst,
                           const float* __restrict__ O, TriP P) {
  __shared__ float xs[2][DP];
  __shared__ float os[2][DH];
  __shared__ float red[2][DP];
  int t = threadIdx.x, sub = t >> 7, p = t & 127;
  long pi = (long)blockIdx.x * 2 + sub;
  float xv = xsrc[pi*DP + p];
  red[sub][p] = xv; __syncthreads();
  for (int s = 64; s > 0; s >>= 1) { if (p < s) red[sub][p] += red[sub][p+s]; __syncthreads(); }
  float mean = red[sub][0] * (1.0f / DP); __syncthreads();
  red[sub][p] = (xv-mean)*(xv-mean); __syncthreads();
  for (int s = 64; s > 0; s >>= 1) { if (p < s) red[sub][p] += red[sub][p+s]; __syncthreads(); }
  float inv = rsqrtf(red[sub][0] * (1.0f / DP) + 1e-5f); __syncthreads();
  xs[sub][p] = (xv-mean)*inv*P.ng[p] + P.nb[p];
  float ov = (p < DH) ? O[pi*DH + p] : 0.f;
  red[sub][p] = (p < DH) ? ov : 0.f; __syncthreads();
  for (int s = 64; s > 0; s >>= 1) { if (p < s) red[sub][p] += red[sub][p+s]; __syncthreads(); }
  float om = red[sub][0] * (1.0f / DH); __syncthreads();
  red[sub][p] = (p < DH) ? (ov-om)*(ov-om) : 0.f; __syncthreads();
  for (int s = 64; s > 0; s >>= 1) { if (p < s) red[sub][p] += red[sub][p+s]; __syncthreads(); }
  float oinv = rsqrtf(red[sub][0] * (1.0f / DH) + 1e-5f); __syncthreads();
  if (p < DH) os[sub][p] = (ov-om)*oinv*P.nog[p] + P.nob[p];
  __syncthreads();
  float g = P.gb[p], y = P.ob[p];
  for (int ch = 0; ch < DP; ++ch) g += xs[sub][ch] * P.gw[ch*DP + p];
  for (int d = 0; d < DH; ++d)   y += os[sub][d] * P.ow[d*DP + p];
  dst[pi*DP + p] = resid[pi*DP + p] + sigm(g) * y;
}

// ---------------------------------------------------------------------------
// Axial attention (tied over n)
// ---------------------------------------------------------------------------
struct AttnP {
  const float *npg,*npb,*nbg,*nbb,*qw,*kw,*vw,*bw,*gw,*gb,*ow,*ob;
};

__global__ void k_attn_qkv(const float* __restrict__ pairbuf, AttnP P, int is_row,
                           us* __restrict__ qA, us* __restrict__ kB, us* __restrict__ vB) {
  __shared__ float xs[DP]; __shared__ float red[DP];
  int t = threadIdx.x;
  long pos = blockIdx.x;
  int n = (int)(pos >> 9), i2 = (int)(pos & 511);
  long si = is_row ? ((long)i2 * L + n) : pos;
  float xv = pairbuf[si*DP + t];
  red[t] = xv; __syncthreads();
  for (int s = 64; s > 0; s >>= 1) { if (t < s) red[t] += red[t+s]; __syncthreads(); }
  float mean = red[0] * (1.0f / DP); __syncthreads();
  red[t] = (xv-mean)*(xv-mean); __syncthreads();
  for (int s = 64; s > 0; s >>= 1) { if (t < s) red[t] += red[t+s]; __syncthreads(); }
  float inv = rsqrtf(red[0] * (1.0f / DP) + 1e-5f); __syncthreads();
  xs[t] = (xv-mean)*inv*P.npg[t] + P.npb[t];
  __syncthreads();
  float q = 0.f, k = 0.f, v = 0.f;
  for (int c = 0; c < DP; ++c) {
    float x = xs[c];
    q += x * P.qw[c*DP + t]; k += x * P.kw[c*DP + t]; v += x * P.vw[c*DP + t];
  }
  q *= 0.17677669529663687f;     // 1/sqrt(32)
  k *= (1.0f / L);               // tied attention norm
  int h = t >> 5, d = t & 31;
  long nd = (long)n * DH + d;
  qA[((long)h * L + i2) * (L*DH) + nd] = f2bf(q);
  kB[((long)h * L + i2) * (L*DH) + nd] = f2bf(k);
  vB[((long)h * (L*DH) + nd) * L + i2] = f2bf(v);
}

__global__ void k_b2(const float* __restrict__ BIAS, AttnP P, int is_row,
                     float* __restrict__ B2) {
  __shared__ float xs[DP]; __shared__ float red[DP];
  int t = threadIdx.x;
  long pi = blockIdx.x;
  int i = (int)(pi >> 9), j = (int)(pi & 511);
  long si = is_row ? ((long)j * L + i) : pi;
  float xv = BIAS[si*DP + t];
  red[t] = xv; __syncthreads();
  for (int s = 64; s > 0; s >>= 1) { if (t < s) red[t] += red[t+s]; __syncthreads(); }
  float mean = red[0] * (1.0f / DP); __syncthreads();
  red[t] = (xv-mean)*(xv-mean); __syncthreads();
  for (int s = 64; s > 0; s >>= 1) { if (t < s) red[t] += red[t+s]; __syncthreads(); }
  float inv = rsqrtf(red[0] * (1.0f / DP) + 1e-5f); __syncthreads();
  xs[t] = (xv-mean)*inv*P.nbg[t] + P.nbb[t];
  __syncthreads();
  if (t < HN) {
    float acc = 0.f;
    for (int c = 0; c < DP; ++c) acc += xs[c] * P.bw[c*HN + t];
    B2[pi*HN + t] = acc;
  }
}

// S[i,j,h] = sum_{n,d} q*k + b2 ; K = 16384, 1x4 N-blocked
__global__ __launch_bounds__(32) void k_attn_logits(const us* __restrict__ qA,
                                                    const us* __restrict__ kB,
                                                    const float* __restrict__ B2,
                                                    float* __restrict__ S) {
  int it = blockIdx.x, jq = blockIdx.y, h = blockIdx.z;
  int lane = threadIdx.x, sel = lane >> 4, mn = lane & 15;
  const us* Ab = qA + ((long)h * L + it*16 + mn) * (L*DH);
  const us* Bb[4];
  for (int p = 0; p < 4; ++p)
    Bb[p] = kB + ((long)h * L + jq*64 + p*16 + mn) * (L*DH);
  v8f acc[4];
  for (int p = 0; p < 4; ++p) for (int r = 0; r < 8; ++r) acc[p][r] = 0.f;
  for (int kc = 0; kc < L*DH; kc += 32) {
    Frag a;
    a.q[0] = *(const v4u*)(Ab + kc + sel*8);
    a.q[1] = *(const v4u*)(Ab + kc + 16 + sel*8);
    __builtin_prefetch(Ab + kc + 64, 0, 1);
    for (int p = 0; p < 4; ++p) {
      Frag b;
      b.q[0] = *(const v4u*)(Bb[p] + kc + sel*8);
      b.q[1] = *(const v4u*)(Bb[p] + kc + 16 + sel*8);
      acc[p] = __builtin_amdgcn_wmma_f32_16x16x32_bf16(false, a.v, false, b.v,
                                                       (short)0, acc[p], false, false);
    }
  }
  for (int p = 0; p < 4; ++p) {
    int j = jq*64 + p*16 + mn;
    for (int r = 0; r < 8; ++r) {
      int i = it*16 + r + sel*8;
      S[(long)h * LL + (long)i * L + j] = acc[p][r] + B2[((long)i * L + j) * HN + h];
    }
  }
}

__global__ void k_softmax(const float* __restrict__ S, us* __restrict__ Pm) {
  __shared__ float red[256];
  int t = threadIdx.x, i = blockIdx.x, h = blockIdx.y;
  const float* row = S + (long)h * LL + (long)i * L;
  float v0 = row[t], v1 = row[t + 256];
  red[t] = fmaxf(v0, v1); __syncthreads();
  for (int s = 128; s > 0; s >>= 1) { if (t < s) red[t] = fmaxf(red[t], red[t+s]); __syncthreads(); }
  float m = red[0]; __syncthreads();
  float e0 = __expf(v0 - m), e1 = __expf(v1 - m);
  red[t] = e0 + e1; __syncthreads();
  for (int s = 128; s > 0; s >>= 1) { if (t < s) red[t] += red[t+s]; __syncthreads(); }
  float sinv = 1.0f / red[0];
  us* prow = Pm + (long)h * LL + (long)i * L;
  prow[t] = f2bf(e0 * sinv); prow[t + 256] = f2bf(e1 * sinv);
}

// O[i, nd] = sum_j P[i,j] * V[nd, j] ; 1x4 N-blocked over nd
__global__ __launch_bounds__(32) void k_attn_out(const us* __restrict__ Pm,
                                                 const us* __restrict__ vB,
                                                 us* __restrict__ Ob) {
  int it = blockIdx.x, ndq = blockIdx.y, h = blockIdx.z;
  int lane = threadIdx.x, sel = lane >> 4, mn = lane & 15;
  const us* Ab = Pm + (long)h * LL + (long)(it*16 + mn) * L;
  const us* Bb[4];
  for (int p = 0; p < 4; ++p)
    Bb[p] = vB + ((long)h * (L*DH) + ndq*64 + p*16 + mn) * L;
  v8f acc[4];
  for (int p = 0; p < 4; ++p) for (int r = 0; r < 8; ++r) acc[p][r] = 0.f;
  for (int kc = 0; kc < L; kc += 32) {
    Frag a;
    a.q[0] = *(const v4u*)(Ab + kc + sel*8);
    a.q[1] = *(const v4u*)(Ab + kc + 16 + sel*8);
    for (int p = 0; p < 4; ++p) {
      Frag b;
      b.q[0] = *(const v4u*)(Bb[p] + kc + sel*8);
      b.q[1] = *(const v4u*)(Bb[p] + kc + 16 + sel*8);
      acc[p] = __builtin_amdgcn_wmma_f32_16x16x32_bf16(false, a.v, false, b.v,
                                                       (short)0, acc[p], false, false);
    }
  }
  for (int p = 0; p < 4; ++p) {
    int nd = ndq*64 + p*16 + mn;
    for (int r = 0; r < 8; ++r) {
      int i = it*16 + r + sel*8;
      Ob[((long)h * L + i) * (L*DH) + nd] = f2bf(acc[p][r]);
    }
  }
}

__global__ void k_attn_post(float* __restrict__ pairbuf, AttnP P, int is_row,
                            const us* __restrict__ Ob) {
  __shared__ float xs[DP]; __shared__ float ts[DP]; __shared__ float red[DP];
  int t = threadIdx.x;
  long pos = blockIdx.x;
  int n = (int)(pos >> 9), i2 = (int)(pos & 511);
  long si = is_row ? ((long)i2 * L + n) : pos;
  float xv = pairbuf[si*DP + t];
  red[t] = xv; __syncthreads();
  for (int s = 64; s > 0; s >>= 1) { if (t < s) red[t] += red[t+s]; __syncthreads(); }
  float mean = red[0] * (1.0f / DP); __syncthreads();
  red[t] = (xv-mean)*(xv-mean); __syncthreads();
  for (int s = 64; s > 0; s >>= 1) { if (t < s) red[t] += red[t+s]; __syncthreads(); }
  float inv = rsqrtf(red[0] * (1.0f / DP) + 1e-5f); __syncthreads();
  xs[t] = (xv-mean)*inv*P.npg[t] + P.npb[t];
  __syncthreads();
  int h = t >> 5, d = t & 31;
  float g = P.gb[t];
  for (int c = 0; c < DP; ++c) g += xs[c] * P.gw[c*DP + t];
  float o = bf2f(Ob[((long)h * L + i2) * (L*DH) + (long)n * DH + d]);
  ts[t] = sigm(g) * o;
  __syncthreads();
  float y = P.ob[t];
  for (int c = 0; c < DP; ++c) y += ts[c] * P.ow[c*DP + t];
  pairbuf[si*DP + t] = xv + y;
}

// ---------------------------------------------------------------------------
// Feed-forward
// ---------------------------------------------------------------------------
__global__ void k_ff(float* __restrict__ pairbuf,
                     const float* __restrict__ ng, const float* __restrict__ nb,
                     const float* __restrict__ w1, const float* __restrict__ b1,
                     const float* __restrict__ w2, const float* __restrict__ b2) {
  __shared__ float xs[DP]; __shared__ float hs[2*DP]; __shared__ float red[256];
  int t = threadIdx.x;
  long pi = blockIdx.x;
  float xv = (t < DP) ? pairbuf[pi*DP + t] : 0.f;
  red[t] = (t < DP) ? xv : 0.f; __syncthreads();
  for (int s = 128; s > 0; s >>= 1) { if (t < s) red[t] += red[t+s]; __syncthreads(); }
  float mean = red[0] * (1.0f / DP); __syncthreads();
  red[t] = (t < DP) ? (xv-mean)*(xv-mean) : 0.f; __syncthreads();
  for (int s = 128; s > 0; s >>= 1) { if (t < s) red[t] += red[t+s]; __syncthreads(); }
  float inv = rsqrtf(red[0] * (1.0f / DP) + 1e-5f); __syncthreads();
  if (t < DP) xs[t] = (xv-mean)*inv*ng[t] + nb[t];
  __syncthreads();
  float hv = b1[t];
  for (int c = 0; c < DP; ++c) hv += xs[c] * w1[c*(2*DP) + t];
  hs[t] = fmaxf(hv, 0.f);
  __syncthreads();
  if (t < DP) {
    float y = b2[t];
    for (int q = 0; q < 2*DP; ++q) y += hs[q] * w2[q*DP + t];
    pairbuf[pi*DP + t] = xv + y;
  }
}

// ---------------------------------------------------------------------------
// Host launch
// ---------------------------------------------------------------------------
static TriP mk_tri(void* const* d, int b) {
  TriP p;
  p.ng=(const float*)d[b+0];  p.nb=(const float*)d[b+1];
  p.lw=(const float*)d[b+2];  p.lb=(const float*)d[b+3];
  p.rw=(const float*)d[b+4];  p.rb=(const float*)d[b+5];
  p.lgw=(const float*)d[b+6]; p.lgb=(const float*)d[b+7];
  p.rgw=(const float*)d[b+8]; p.rgb=(const float*)d[b+9];
  p.gw=(const float*)d[b+10]; p.gb=(const float*)d[b+11];
  p.nog=(const float*)d[b+12];p.nob=(const float*)d[b+13];
  p.ow=(const float*)d[b+14]; p.ob=(const float*)d[b+15];
  return p;
}
static AttnP mk_attn(void* const* d, int b) {
  AttnP p;
  p.npg=(const float*)d[b+0]; p.npb=(const float*)d[b+1];
  p.nbg=(const float*)d[b+2]; p.nbb=(const float*)d[b+3];
  p.qw=(const float*)d[b+4];  p.kw=(const float*)d[b+5];
  p.vw=(const float*)d[b+6];  p.bw=(const float*)d[b+7];
  p.gw=(const float*)d[b+8];  p.gb=(const float*)d[b+9];
  p.ow=(const float*)d[b+10]; p.ob=(const float*)d[b+11];
  return p;
}

extern "C" void kernel_launch(void* const* d_in, const int* in_sizes, int n_in,
                              void* d_out, int out_size, void* d_ws, size_t ws_size,
                              hipStream_t stream) {
  (void)in_sizes; (void)n_in; (void)out_size; (void)ws_size;
  const float* pair  = (const float*)d_in[0];
  const float* rbf   = (const float*)d_in[1];
  const float* state = (const float*)d_in[2];
  float* out = (float*)d_out;

  unsigned char* ws = (unsigned char*)d_ws;
  float* BIAS = (float*)ws;                                     // 134.2 MB
  float* LR   = (float*)(ws + (size_t)LL * DP * 4);             // 64 KB
  unsigned char* SCR = ws + (size_t)LL * DP * 4 + 65536;        // aliased region

  // triangle staging (within SCR)
  us*    triA = (us*)SCR;                                       // 16.8 MB
  us*    triB = triA + (size_t)DH * LL;                         // 16.8 MB
  float* triO = (float*)(SCR + (size_t)DH * LL * 2 * 2);        // 33.6 MB

  // attention staging (reuses SCR after triangles)
  size_t QE = (size_t)HN * L * L * DH;                          // 33.5M elems
  us*    qA  = (us*)SCR;
  us*    kBf = qA + QE;
  us*    vBf = kBf + QE;
  us*    Ob  = vBf + QE;
  float* Sl  = (float*)(Ob + QE);
  us*    Pm  = (us*)((unsigned char*)Sl + (size_t)HN * LL * 4);
  float* B2  = (float*)((unsigned char*)Pm + (size_t)HN * LL * 2);

  // stage 0/1: state proj + gated RBF bias tensor
  k_state_proj<<<2, 256, 0, stream>>>(state,
      (const float*)d_in[3], (const float*)d_in[4],
      (const float*)d_in[5], (const float*)d_in[6],
      (const float*)d_in[7], (const float*)d_in[8], LR);
  k_bias<<<LL/2, 256, 0, stream>>>(rbf, LR,
      (const float*)d_in[9], (const float*)d_in[10],
      (const float*)d_in[11], (const float*)d_in[12], BIAS);

  // triangle mult (outgoing): pair_in -> out
  TriP tOut = mk_tri(d_in, 13), tIn = mk_tri(d_in, 29);
  k_tri_pre<<<LL/4, 256, 0, stream>>>(pair, tOut, triA, triB, 0);
  k_tri_mm<<<dim3(32, 8, 32), 32, 0, stream>>>(triA, triB, triO);
  k_tri_post<<<LL/2, 256, 0, stream>>>(pair, pair, out, triO, tOut);

  // triangle mult (incoming): in-place on out
  k_tri_pre<<<LL/4, 256, 0, stream>>>(out, tIn, triA, triB, 1);
  k_tri_mm<<<dim3(32, 8, 32), 32, 0, stream>>>(triA, triB, triO);
  k_tri_post<<<LL/2, 256, 0, stream>>>(out, out, out, triO, tIn);

  // row attention (transposed axes), then column attention
  AttnP aRow = mk_attn(d_in, 45), aCol = mk_attn(d_in, 57);
  for (int pass = 0; pass < 2; ++pass) {
    AttnP P = pass == 0 ? aRow : aCol;
    int is_row = pass == 0 ? 1 : 0;
    k_attn_qkv<<<LL, 128, 0, stream>>>(out, P, is_row, qA, kBf, vBf);
    k_b2<<<LL, 128, 0, stream>>>(BIAS, P, is_row, B2);
    k_attn_logits<<<dim3(32, 8, HN), 32, 0, stream>>>(qA, kBf, B2, Sl);
    k_softmax<<<dim3(L, HN), 256, 0, stream>>>(Sl, Pm);
    k_attn_out<<<dim3(32, 256, HN), 32, 0, stream>>>(Pm, vBf, Ob);
    k_attn_post<<<LL, 128, 0, stream>>>(out, P, is_row, Ob);
  }

  // feed-forward (in-place)
  k_ff<<<LL, 256, 0, stream>>>(out,
      (const float*)d_in[69], (const float*)d_in[70],
      (const float*)d_in[71], (const float*)d_in[72],
      (const float*)d_in[73], (const float*)d_in[74]);
}
